// PrepareForMultiHeadAttention_31971736551588
// MI455X (gfx1250) — matrix operand order repaired and checked by reading, not compiled, not run on hardware
//
#include <hip/hip_runtime.h>

// ---------------------------------------------------------------------------
// Per-position branched GEMM:  out[b,s,:] = x[b,s,:] @ W[idx(s)]^T + sum_j b[j]
// B=128, S=513, H=256, fp32 in/out.
// bf16 split-precision (hi+lo) -> 3x v_wmma_f32_16x16x32_bf16, fp32 accum.
// One workgroup = one position s x one 128-wide N half.
// Round 3: B-panel staging via global_load_async_to_lds_b64 (ASYNCcnt),
// overlapped with the VALU bf16-split of the A panel.
// ---------------------------------------------------------------------------

typedef __attribute__((ext_vector_type(16))) __bf16 v16bf;
typedef __attribute__((ext_vector_type(2)))  __bf16 bf16x2;
typedef __attribute__((ext_vector_type(8)))  float  v8f;

#define TILE_K     128
#define LDS_PITCH  136   // 128 + 8 halfword pad: 272B rows -> 68-dword stride,
                         // conflict-free b128 fragment loads across 64 banks

union FragU {
    v16bf v;
    uint4 q[2];
};

// Split x into bf16 hi + bf16 lo using the hardware converter (RNE).
__device__ __forceinline__ void split_pk(float x0, float x1,
                                         unsigned &hi, unsigned &lo) {
    bf16x2 h = { (__bf16)x0, (__bf16)x1 };
    bf16x2 l = { (__bf16)(x0 - (float)h[0]), (__bf16)(x1 - (float)h[1]) };
    hi = __builtin_bit_cast(unsigned, h);
    lo = __builtin_bit_cast(unsigned, l);
}

// ---- pre-pass: split W[5,H,H] fp32 into Whi/Wlo bf16 once -----------------
__global__ __launch_bounds__(256) void split_weights(
    const float* __restrict__ W,
    unsigned short* __restrict__ Whi,
    unsigned short* __restrict__ Wlo,
    int n)   // total elements, even
{
    int i2 = (blockIdx.x * 256 + threadIdx.x) * 2;
    if (i2 >= n) return;
    unsigned h, l;
    split_pk(W[i2], W[i2 + 1], h, l);
    *reinterpret_cast<unsigned*>(&Whi[i2]) = h;
    *reinterpret_cast<unsigned*>(&Wlo[i2]) = l;
}

// ---- main kernel ----------------------------------------------------------
template <bool PRECONV>
__global__ __launch_bounds__(256) void branch_gemm_bf16x3(
    const float* __restrict__ x,      // [B,S,H]
    const float* __restrict__ W,      // [5,H,H]
    const float* __restrict__ bias5,  // [5,H]
    const unsigned short* __restrict__ Whi,  // [5,H,H] bf16 (if PRECONV)
    const unsigned short* __restrict__ Wlo,  // [5,H,H] bf16 (if PRECONV)
    float* __restrict__ out,          // [B,S,H]
    int S, int H)
{
    const int s      = blockIdx.x;        // sequence position
    const int n_base = blockIdx.y * 128;  // output-column half

    // branch index: 0,1,2 then alternating 3/4
    const int idx = (s < 3) ? s : (3 + ((s - 3) & 1));
    const size_t wbase = (size_t)idx * H * H;

    __shared__ __align__(16) unsigned short sAhi[128 * LDS_PITCH];
    __shared__ __align__(16) unsigned short sAlo[128 * LDS_PITCH];
    __shared__ __align__(16) unsigned short sBhi[128 * LDS_PITCH];
    __shared__ __align__(16) unsigned short sBlo[128 * LDS_PITCH];
    __shared__ float sBias[128];

    const int tid  = threadIdx.x;
    const int lane = tid & 31;
    const int wave = tid >> 5;
    const int wm   = wave >> 1;   // 0..3 : M block of 32 (batch rows)
    const int wn   = wave & 1;    // 0..1 : N block of 64 (out cols)
    const int half = lane >> 4;
    const int l15  = lane & 15;

    if (tid < 128) {
        float a = 0.f;
        #pragma unroll
        for (int j = 0; j < 5; ++j) a += bias5[j * H + n_base + tid];
        sBias[tid] = a;
    }

    v8f acc[2][4];
    #pragma unroll
    for (int mi = 0; mi < 2; ++mi)
        #pragma unroll
        for (int ni = 0; ni < 4; ++ni)
            acc[mi][ni] = v8f{};

    const size_t rowStrideX = (size_t)S * H;
    const float* xs = x + (size_t)s * H;

    // Uniform 64-bit bases for async B staging (SGPR operands).
    const unsigned long long whiBase =
        (unsigned long long)(uintptr_t)(Whi + wbase);
    const unsigned long long wloBase =
        (unsigned long long)(uintptr_t)(Wlo + wbase);

    for (int kc = 0; kc < 2; ++kc) {
        const int k0 = kc * TILE_K;

        // ---- staging ----
        // 1) B panel: fire-and-forget async DMA global->LDS (no VGPR data),
        //    runs concurrently with the A-panel split below.
        if (PRECONV) {
            #pragma unroll
            for (int i = 0; i < 16; ++i) {
                const int li  = i * 256 + tid;
                const int row = li >> 5;
                const int c4  = li & 31;
                const int off = row * LDS_PITCH + c4 * 4;
                const unsigned ldsBhi = (unsigned)(uintptr_t)&sBhi[off];
                const unsigned ldsBlo = (unsigned)(uintptr_t)&sBlo[off];
                const unsigned voff =
                    (unsigned)(((n_base + row) * H + k0 + c4 * 4) * 2); // bytes
                asm volatile("global_load_async_to_lds_b64 %0, %1, %2"
                             :: "v"(ldsBhi), "v"(voff), "s"(whiBase)
                             : "memory");
                asm volatile("global_load_async_to_lds_b64 %0, %1, %2"
                             :: "v"(ldsBlo), "v"(voff), "s"(wloBase)
                             : "memory");
            }
        }

        // 2) A panel (and B panel in fallback): load fp32, hw-cvt split, store.
        #pragma unroll
        for (int i = 0; i < 16; ++i) {
            const int li  = i * 256 + tid;
            const int row = li >> 5;
            const int c4  = li & 31;
            const int off = row * LDS_PITCH + c4 * 4;   // halfwords, 8B aligned

            const float4 a4 = *reinterpret_cast<const float4*>(
                xs + (size_t)row * rowStrideX + k0 + c4 * 4);
            uint2 th, tl;
            split_pk(a4.x, a4.y, th.x, tl.x);
            split_pk(a4.z, a4.w, th.y, tl.y);
            *reinterpret_cast<uint2*>(&sAhi[off]) = th;
            *reinterpret_cast<uint2*>(&sAlo[off]) = tl;

            if (!PRECONV) {
                const size_t woff =
                    wbase + (size_t)(n_base + row) * H + k0 + c4 * 4;
                const float4 b4 = *reinterpret_cast<const float4*>(W + woff);
                uint2 bh, bl;
                split_pk(b4.x, b4.y, bh.x, bl.x);
                split_pk(b4.z, b4.w, bh.y, bl.y);
                *reinterpret_cast<uint2*>(&sBhi[off]) = bh;
                *reinterpret_cast<uint2*>(&sBlo[off]) = bl;
            }
        }
        if (PRECONV) {
            asm volatile("s_wait_asynccnt 0x0" ::: "memory");
        }
        __syncthreads();

        // ---- compute: K in steps of 32 ----
        #pragma unroll
        for (int kk = 0; kk < TILE_K; kk += 32) {
            // A frags (16x32 bf16): lane half -> K {0..7,16..23} vs {8..15,24..31}
            FragU ahiF[2], aloF[2];
            #pragma unroll
            for (int mi = 0; mi < 2; ++mi) {
                const int aoff = (wm * 32 + mi * 16 + l15) * LDS_PITCH + kk + half * 8;
                ahiF[mi].q[0] = *reinterpret_cast<const uint4*>(&sAhi[aoff]);
                ahiF[mi].q[1] = *reinterpret_cast<const uint4*>(&sAhi[aoff + 16]);
                aloF[mi].q[0] = *reinterpret_cast<const uint4*>(&sAlo[aoff]);
                aloF[mi].q[1] = *reinterpret_cast<const uint4*>(&sAlo[aoff + 16]);
            }
            #pragma unroll
            for (int ni = 0; ni < 4; ++ni) {
                // B frags (32x16 bf16): lane half -> K {0..7,16..23} vs {8..15,24..31}
                const int boff = (wn * 64 + ni * 16 + l15) * LDS_PITCH + kk + half * 8;
                FragU bhiF, bloF;
                bhiF.q[0] = *reinterpret_cast<const uint4*>(&sBhi[boff]);
                bhiF.q[1] = *reinterpret_cast<const uint4*>(&sBhi[boff + 16]);
                bloF.q[0] = *reinterpret_cast<const uint4*>(&sBlo[boff]);
                bloF.q[1] = *reinterpret_cast<const uint4*>(&sBlo[boff + 16]);
                #pragma unroll
                for (int mi = 0; mi < 2; ++mi) {
                    acc[mi][ni] = __builtin_amdgcn_wmma_f32_16x16x32_bf16(
                        false, ahiF[mi].v, false, bhiF.v, (short)0, acc[mi][ni], false, false);
                    acc[mi][ni] = __builtin_amdgcn_wmma_f32_16x16x32_bf16(
                        false, ahiF[mi].v, false, bloF.v, (short)0, acc[mi][ni], false, false);
                    acc[mi][ni] = __builtin_amdgcn_wmma_f32_16x16x32_bf16(
                        false, aloF[mi].v, false, bhiF.v, (short)0, acc[mi][ni], false, false);
                }
            }
        }
        __syncthreads();
    }

    // ---- epilogue: C/D layout -> VGPR i holds M = half*8 + i, N = l15 ----
    #pragma unroll
    for (int mi = 0; mi < 2; ++mi) {
        #pragma unroll
        for (int ni = 0; ni < 4; ++ni) {
            const int nloc = wn * 64 + ni * 16 + l15;
            const int d    = n_base + nloc;
            const float bs = sBias[nloc];
            #pragma unroll
            for (int i = 0; i < 8; ++i) {
                const int brow = wm * 32 + mi * 16 + half * 8 + i;  // batch row
                out[((size_t)brow * S + s) * H + d] = acc[mi][ni][i] + bs;
            }
        }
    }
}

extern "C" void kernel_launch(void* const* d_in, const int* in_sizes, int n_in,
                              void* d_out, int out_size, void* d_ws, size_t ws_size,
                              hipStream_t stream) {
    const float* x = (const float*)d_in[0];   // [128,513,256]
    const float* W = (const float*)d_in[1];   // [5,256,256]
    const float* b = (const float*)d_in[2];   // [5,256]
    float* out = (float*)d_out;               // [128,513,4,64] flat

    const int S = 513, H = 256;
    const size_t nW = (size_t)5 * H * H;      // 327680 elements

    dim3 grid(S, 2, 1);
    dim3 block(256, 1, 1);

    if (ws_size >= nW * 2 * sizeof(unsigned short)) {
        unsigned short* Whi = (unsigned short*)d_ws;
        unsigned short* Wlo = Whi + nW;
        const int nThreads = (int)(nW / 2);   // 2 elements per thread
        split_weights<<<(nThreads + 255) / 256, 256, 0, stream>>>(W, Whi, Wlo, (int)nW);
        branch_gemm_bf16x3<true><<<grid, block, 0, stream>>>(x, W, b, Whi, Wlo, out, S, H);
    } else {
        branch_gemm_bf16x3<false><<<grid, block, 0, stream>>>(x, W, b, nullptr, nullptr, out, S, H);
    }
}
